// LSTMTagger_12146167513228
// MI455X (gfx1250) — matrix operand order repaired and checked by reading, not compiled, not run on hardware
//
#include <hip/hip_runtime.h>

// ---------------- problem dims ----------------
#define NCC   8192      // n chars
#define NWW   2048      // n words
#define ECD   128       // char emb dim
#define EDD   1024      // word emb dim
#define HCD   256       // char hidden
#define HDD   1024      // word hidden
#define XDD   1280      // EDD + HCD
#define TDD   64        // tagset
#define NWGW  32        // workgroups for word recurrence

typedef __attribute__((ext_vector_type(16))) __bf16 v16bf;
typedef __attribute__((ext_vector_type(8)))  float  v8f;

// ---- bf16 helpers (manual RNE packing; decode via shift) ----
__device__ __forceinline__ unsigned short f2bf(float f) {
    unsigned int u = __float_as_uint(f);
    unsigned int r = u + 0x7FFFu + ((u >> 16) & 1u);
    return (unsigned short)(r >> 16);
}
__device__ __forceinline__ float bf2f(unsigned int b) {
    return __uint_as_float((b & 0xFFFFu) << 16);
}

// Load 16 bf16 for the A operand: halves {p[0..7], p[16..23]} (ISA 7.12.2
// 16-bit A 16x32 layout: each lane-half carries K 0..7 and 16..23 rel. base).
__device__ __forceinline__ v16bf ld_a16(const unsigned short* p) {
    union { uint4 q[2]; v16bf v; } u;
    u.q[0] = *(const uint4*)(p);
    u.q[1] = *(const uint4*)(p + 16);
    return u.v;
}
// Load 16 contiguous bf16 (p[0..15]); unused: the GEMM loads B with ld_a16,
// so B uses the same k order as A.
__device__ __forceinline__ v16bf ld_b16(const unsigned short* p) {
    union { uint4 q[2]; v16bf v; } u;
    u.q[0] = *(const uint4*)(p);
    u.q[1] = *(const uint4*)(p + 8);
    return u.v;
}

// =====================================================================
// fp32 -> bf16 conversion (pairs packed into dword stores)
// =====================================================================
__global__ void k_f2bf(const float* __restrict__ in, unsigned int* __restrict__ out, long npairs)
{
    long i = (long)blockIdx.x * blockDim.x + threadIdx.x;
    if (i >= npairs) return;
    const float* p = in + 2 * i;
    out[i] = (unsigned int)f2bf(p[0]) | ((unsigned int)f2bf(p[1]) << 16);
}

// =====================================================================
// GEMM: out[M,N] = gather(A)[M,K](bf16) * W[N,K](bf16)^T + b0 + b1
// One wave per 16x64 strip: 4 accumulators share one A fetch per K-step.
// Inner loop: ~10 b128 loads + 4 v_wmma_f32_16x16x32_bf16, no cvt VALU.
// =====================================================================
__global__ void k_gemm_bf16(const unsigned short* __restrict__ A,
                            const int*   __restrict__ rowIdx,
                            int lda,
                            const unsigned short* __restrict__ W, int ldw,
                            const float* __restrict__ b0,
                            const float* __restrict__ b1,
                            float* __restrict__ out,
                            int Kd, int ldo)
{
    const int lane = threadIdx.x & 31;
    const int n0 = blockIdx.x * 64;
    const int m0 = blockIdx.y * 16;
    const int r  = lane & 15;
    const bool lo = (lane < 16);

    int arow = m0 + r;
    if (rowIdx) arow = rowIdx[arow];
    const unsigned short* pa = A + (long)arow * lda + (lo ? 0 : 8);

    const unsigned short* pb[4];
    #pragma unroll
    for (int nt = 0; nt < 4; ++nt)
        pb[nt] = W + (long)(n0 + nt * 16 + r) * ldw + (lo ? 0 : 8);

    v8f acc0 = {}, acc1 = {}, acc2 = {}, acc3 = {};
    for (int k0 = 0; k0 < Kd; k0 += 32) {
        v16bf av = ld_a16(pa + k0);
        v16bf bv0 = ld_a16(pb[0] + k0);
        v16bf bv1 = ld_a16(pb[1] + k0);
        v16bf bv2 = ld_a16(pb[2] + k0);
        v16bf bv3 = ld_a16(pb[3] + k0);
        acc0 = __builtin_amdgcn_wmma_f32_16x16x32_bf16(false, av, false, bv0, (short)0, acc0, false, false);
        acc1 = __builtin_amdgcn_wmma_f32_16x16x32_bf16(false, av, false, bv1, (short)0, acc1, false, false);
        acc2 = __builtin_amdgcn_wmma_f32_16x16x32_bf16(false, av, false, bv2, (short)0, acc2, false, false);
        acc3 = __builtin_amdgcn_wmma_f32_16x16x32_bf16(false, av, false, bv3, (short)0, acc3, false, false);
    }

    const int mbase = m0 + (lo ? 0 : 8);
    #pragma unroll
    for (int nt = 0; nt < 4; ++nt) {
        const v8f a = (nt == 0) ? acc0 : (nt == 1) ? acc1 : (nt == 2) ? acc2 : acc3;
        const int ncol = n0 + nt * 16 + r;
        float bias = (b0 ? b0[ncol] : 0.f) + (b1 ? b1[ncol] : 0.f);
        #pragma unroll
        for (int rr = 0; rr < 8; ++rr)
            out[(long)(mbase + rr) * ldo + ncol] = a[rr] + bias;
    }
}

// =====================================================================
// Char LSTM recurrence: 1 workgroup, 1024 threads, thread j owns gate
// row j. W_hh_c row held register-resident as packed bf16 (128 dwords).
// h lives in LDS; c owned by threads j<256.
// =====================================================================
__global__ void __launch_bounds__(1024)
k_char_lstm(const float* __restrict__ pre,
            const float* __restrict__ Whh,
            float* __restrict__ hout)
{
    __shared__ float hs[HCD];
    __shared__ float gs[4 * HCD];
    const int j = threadIdx.x;

    unsigned int w[128];
    const float* wr = Whh + (long)j * HCD;
    #pragma unroll
    for (int p = 0; p < 128; ++p)
        w[p] = (unsigned int)f2bf(wr[2*p]) | ((unsigned int)f2bf(wr[2*p+1]) << 16);

    float c = 0.f;
    if (j < HCD) hs[j] = 0.f;
    __syncthreads();

    const int gate = j >> 8;
    for (int t = 0; t < NCC; ++t) {
        float acc = pre[(long)t * (4 * HCD) + j];
        const float4* h4 = (const float4*)hs;
        #pragma unroll 8
        for (int p8 = 0; p8 < 32; ++p8) {
            float4 ha = h4[2*p8], hb = h4[2*p8+1];
            unsigned int w0 = w[4*p8+0], w1 = w[4*p8+1];
            unsigned int w2 = w[4*p8+2], w3 = w[4*p8+3];
            acc += bf2f(w0)*ha.x + bf2f(w0>>16)*ha.y
                 + bf2f(w1)*ha.z + bf2f(w1>>16)*ha.w
                 + bf2f(w2)*hb.x + bf2f(w2>>16)*hb.y
                 + bf2f(w3)*hb.z + bf2f(w3>>16)*hb.w;
        }
        float gv = (gate == 2) ? tanhf(acc) : 1.f / (1.f + __expf(-acc));
        gs[j] = gv;
        __syncthreads();
        if (j < HCD) {
            float iv = gs[j], fv = gs[j+256], gg = gs[j+512], ov = gs[j+768];
            c = fv * c + iv * gg;
            float h = ov * tanhf(c);
            hs[j] = h;
            hout[(long)t * HCD + j] = h;
        }
        __syncthreads();
    }
}

// =====================================================================
// Build word-LSTM input in bf16: xw[n] = [ emb[ix[n]] ; hc_out[offs[n]] ]
// Two elements per thread, packed dword stores.
// =====================================================================
__global__ void k_concat_bf(const float* __restrict__ emb, const int* __restrict__ ix,
                            const float* __restrict__ hc,  const int* __restrict__ offs,
                            unsigned int* __restrict__ xw)
{
    long i = (long)blockIdx.x * blockDim.x + threadIdx.x;   // pair index
    if (i >= (long)NWW * (XDD / 2)) return;
    int n  = (int)(i / (XDD / 2));
    int ep = (int)(i % (XDD / 2)) * 2;
    float v0, v1;
    if (ep < EDD) {
        const float* s = emb + (long)ix[n] * EDD + ep;
        v0 = s[0]; v1 = s[1];
    } else {
        const float* s = hc + (long)offs[n] * HCD + (ep - EDD);
        v0 = s[0]; v1 = s[1];
    }
    xw[i] = (unsigned int)f2bf(v0) | ((unsigned int)f2bf(v1) << 16);
}

// =====================================================================
// Word LSTM recurrence, sharded across 32 WGs (32 hidden units each).
// Each WG caches its 128x1024 W_hh slice in LDS as bf16 (~257KB of the
// 320KB WGP LDS); h exchanged via L2 + device-scope counter sync.
// h output written in bf16 for the tag GEMM.
// =====================================================================
__global__ void __launch_bounds__(128)
k_word_lstm(const float* __restrict__ pre,
            const float* __restrict__ Whh,
            float* __restrict__ hglob,
            int*   __restrict__ cnt,
            unsigned short* __restrict__ hout)
{
    extern __shared__ char smem[];
    const int SW = 1028;                                   // halves per row (8B aligned, low-conflict)
    unsigned short* lw  = (unsigned short*)smem;           // [128][SW]
    float*          hsh = (float*)(smem + 128 * SW * 2);   // 1024 floats
    float*          gsh = (float*)(smem + 128 * SW * 2 + HDD * 4); // 128 floats

    const int wg   = blockIdx.x;
    const int t    = threadIdx.x;
    const int gate = t >> 5;
    const int unit = wg * 32 + (t & 31);
    const int row  = gate * HDD + unit;

    const float* wr = Whh + (long)row * HDD;
    for (int k = 0; k < HDD; ++k) lw[t * SW + k] = f2bf(wr[k]);
    float c = 0.f;
    __syncthreads();

    for (int step = 0; step < NWW; ++step) {
        const float* hsrc = hglob + (step & 1) * HDD;
        for (int k = t; k < HDD; k += 128) hsh[k] = hsrc[k];
        __syncthreads();

        __builtin_prefetch(&pre[(long)(step + 1) * (4 * HDD) + row], 0, 0);
        float acc = pre[(long)step * (4 * HDD) + row];
        const unsigned short* wrow = lw + t * SW;
        const float4* h4 = (const float4*)hsh;
        for (int k8 = 0; k8 < 128; ++k8) {
            uint2 wa = *(const uint2*)(wrow + k8 * 8);
            uint2 wb = *(const uint2*)(wrow + k8 * 8 + 4);
            float4 ha = h4[2*k8], hb = h4[2*k8+1];
            acc += bf2f(wa.x)*ha.x + bf2f(wa.x>>16)*ha.y
                 + bf2f(wa.y)*ha.z + bf2f(wa.y>>16)*ha.w
                 + bf2f(wb.x)*hb.x + bf2f(wb.x>>16)*hb.y
                 + bf2f(wb.y)*hb.z + bf2f(wb.y>>16)*hb.w;
        }
        float gv = (gate == 2) ? tanhf(acc) : 1.f / (1.f + __expf(-acc));
        gsh[t] = gv;
        __syncthreads();
        if (t < 32) {
            float iv = gsh[t], fv = gsh[t+32], gg = gsh[t+64], ov = gsh[t+96];
            c = fv * c + iv * gg;
            float h = ov * tanhf(c);
            hglob[((step + 1) & 1) * HDD + unit] = h;
            hout[(long)step * HDD + unit] = f2bf(h);
        }
        __syncthreads();
        if (t == 0) {
            __hip_atomic_fetch_add(&cnt[step], 1, __ATOMIC_RELEASE, __HIP_MEMORY_SCOPE_AGENT);
            while (__hip_atomic_load(&cnt[step], __ATOMIC_ACQUIRE, __HIP_MEMORY_SCOPE_AGENT) < NWGW) { }
        }
        __syncthreads();
    }
}

// =====================================================================
// log_softmax over 64-wide rows; one wave per row, shuffle reductions.
// =====================================================================
__global__ void k_logsoftmax(const float* __restrict__ logits, float* __restrict__ out)
{
    const int lane = threadIdx.x & 31;
    const int wid  = threadIdx.x >> 5;
    const int row  = blockIdx.x * 8 + wid;
    const float* p = logits + (long)row * TDD;
    float a = p[lane], b = p[lane + 32];
    float m = fmaxf(a, b);
    for (int s = 16; s; s >>= 1) m = fmaxf(m, __shfl_xor(m, s, 32));
    float e = __expf(a - m) + __expf(b - m);
    for (int s = 16; s; s >>= 1) e += __shfl_xor(e, s, 32);
    float lse = m + __logf(e);
    out[(long)row * TDD + lane]      = a - lse;
    out[(long)row * TDD + lane + 32] = b - lse;
}

__global__ void k_init(float* hglob, int* cnt)
{
    int i = blockIdx.x * blockDim.x + threadIdx.x;
    if (i < 2 * HDD) hglob[i] = 0.f;
    if (i < NWW)     cnt[i]   = 0;
}

// =====================================================================
extern "C" void kernel_launch(void* const* d_in, const int* in_sizes, int n_in,
                              void* d_out, int out_size, void* d_ws, size_t ws_size,
                              hipStream_t stream)
{
    const int*   ix_c = (const int*)  d_in[0];
    const int*   ix_w = (const int*)  d_in[1];
    const int*   offs = (const int*)  d_in[2];
    const float* cemb = (const float*)d_in[3];
    const float* emb  = (const float*)d_in[4];
    const float* Wihc = (const float*)d_in[5];
    const float* Whhc = (const float*)d_in[6];
    const float* bihc = (const float*)d_in[7];
    const float* bhhc = (const float*)d_in[8];
    const float* Wih  = (const float*)d_in[9];
    const float* Whh  = (const float*)d_in[10];
    const float* bih  = (const float*)d_in[11];
    const float* bhh  = (const float*)d_in[12];
    const float* Wtag = (const float*)d_in[13];
    const float* btag = (const float*)d_in[14];

    char* base = (char*)d_ws;
    auto alloc = [&](size_t bytes) -> char* {
        char* p = base; base += (bytes + 255) & ~(size_t)255; return p;
    };
    float* pre_c   = (float*)alloc((size_t)NCC * 4 * HCD * 4);
    float* hcout   = (float*)alloc((size_t)NCC * HCD * 4);
    float* pre_w   = (float*)alloc((size_t)NWW * 4 * HDD * 4);
    float* logits  = (float*)alloc((size_t)NWW * TDD * 4);
    float* hglob   = (float*)alloc((size_t)2 * HDD * 4);
    int*   cnt     = (int*)  alloc((size_t)NWW * 4);
    unsigned short* cemb_bf = (unsigned short*)alloc((size_t)256 * ECD * 2);
    unsigned short* Wihc_bf = (unsigned short*)alloc((size_t)4 * HCD * ECD * 2);
    unsigned short* Wih_bf  = (unsigned short*)alloc((size_t)4 * HDD * XDD * 2);
    unsigned short* Wtag_bf = (unsigned short*)alloc((size_t)TDD * HDD * 2);
    unsigned short* xw_bf   = (unsigned short*)alloc((size_t)NWW * XDD * 2);
    unsigned short* hw_bf   = (unsigned short*)alloc((size_t)NWW * HDD * 2);

    k_init<<<dim3((NWW + 255) / 256), 256, 0, stream>>>(hglob, cnt);

    // one-shot bf16 conversions of reused operands
    {
        long np;
        np = (long)256 * ECD / 2;
        k_f2bf<<<(int)((np + 255) / 256), 256, 0, stream>>>(cemb, (unsigned int*)cemb_bf, np);
        np = (long)4 * HCD * ECD / 2;
        k_f2bf<<<(int)((np + 255) / 256), 256, 0, stream>>>(Wihc, (unsigned int*)Wihc_bf, np);
        np = (long)4 * HDD * XDD / 2;
        k_f2bf<<<(int)((np + 255) / 256), 256, 0, stream>>>(Wih, (unsigned int*)Wih_bf, np);
        np = (long)TDD * HDD / 2;
        k_f2bf<<<(int)((np + 255) / 256), 256, 0, stream>>>(Wtag, (unsigned int*)Wtag_bf, np);
    }

    // char pre-activations: gather(char_emb) @ W_ih_c^T + b_ih_c + b_hh_c
    k_gemm_bf16<<<dim3((4 * HCD) / 64, NCC / 16), 32, 0, stream>>>(
        cemb_bf, ix_c, ECD, Wihc_bf, ECD, bihc, bhhc, pre_c, ECD, 4 * HCD);

    k_char_lstm<<<1, 1024, 0, stream>>>(pre_c, Whhc, hcout);

    k_concat_bf<<<(int)(((long)NWW * (XDD / 2) + 255) / 256), 256, 0, stream>>>(
        emb, ix_w, hcout, offs, (unsigned int*)xw_bf);

    // word pre-activations: xw @ W_ih^T + b_ih + b_hh
    k_gemm_bf16<<<dim3((4 * HDD) / 64, NWW / 16), 32, 0, stream>>>(
        xw_bf, nullptr, XDD, Wih_bf, XDD, bih, bhh, pre_w, XDD, 4 * HDD);

    size_t lds = (size_t)128 * 1028 * 2 + HDD * 4 + 128 * 4;
    k_word_lstm<<<NWGW, 128, lds, stream>>>(pre_w, Whh, hglob, cnt, hw_bf);

    // tag logits
    k_gemm_bf16<<<dim3(TDD / 64, NWW / 16), 32, 0, stream>>>(
        hw_bf, nullptr, HDD, Wtag_bf, HDD, btag, nullptr, logits, HDD, TDD);

    k_logsoftmax<<<NWW / 8, 256, 0, stream>>>(logits, (float*)d_out);
}
